// Multi_head_temporal_attention_88055419502893
// MI455X (gfx1250) — matrix operand order repaired and checked by reading, not compiled, not run on hardware
//
#include <hip/hip_runtime.h>
#include <hip/hip_bf16.h>

// ---------------------------------------------------------------------------
// Problem constants (reference): B=4, T=12, N=207, D=64, H=8, DH=8
// tokens = B*T*N = 9936 (621 tiles of 16)
// ---------------------------------------------------------------------------
#define BB 4
#define TT 12
#define NN 207
#define DD 64
#define HH 8
#define DH 8
#define NTOK (BB*TT*NN)        // 9936
#define NTILE (NTOK/16)        // 621
#define KDIM (DD*DD)           // 4096
#define M2ELEMS (DD*KDIM)      // 262144 per branch

typedef __attribute__((ext_vector_type(16))) __bf16 v16bf;
typedef __attribute__((ext_vector_type(8)))  __bf16 v8bf;
typedef __attribute__((ext_vector_type(8)))  float  v8f;

__device__ __forceinline__ v16bf cat16(v8bf lo, v8bf hi) {
    v16bf r;
#pragma unroll
    for (int j = 0; j < 8; ++j) { r[j] = lo[j]; r[j + 8] = hi[j]; }
    return r;
}

// ---------------------------------------------------------------------------
// Kernel 0: transpose+convert weights to bf16.
//   Bt [i][e][k] = meta2_w[i][k*64+e]   (k = c*64+d)  : 3*262144
//   W1t[i][e][d] = meta1_w[i][d*64+e]                 : 3*4096
//   B2t[i][e][d] = meta2_b[i][d*64+e]  (b2 as 64x64)  : 3*4096
// ---------------------------------------------------------------------------
__global__ __launch_bounds__(256)
void convert_kernel(const float* __restrict__ m2w, const float* __restrict__ m1w,
                    const float* __restrict__ m2b,
                    __bf16* __restrict__ Bt, __bf16* __restrict__ W1t,
                    __bf16* __restrict__ B2t) {
    int flat0 = (blockIdx.x * 256 + threadIdx.x) * 8;   // total 811008
#pragma unroll
    for (int u = 0; u < 8; ++u) {
        int flat = flat0 + u;
        if (flat < 3 * M2ELEMS) {
            int i = flat >> 18;
            int rem = flat & (M2ELEMS - 1);
            int e = rem >> 12;
            int k = rem & (KDIM - 1);
            Bt[flat] = (__bf16)m2w[i * M2ELEMS + k * DD + e];
        } else if (flat < 3 * M2ELEMS + 3 * 4096) {
            int f = flat - 3 * M2ELEMS;
            int i = f >> 12;
            int rem = f & 4095;
            int e = rem >> 6, d = rem & 63;
            W1t[f] = (__bf16)m1w[i * 4096 + d * DD + e];
        } else {
            int f = flat - 3 * M2ELEMS - 3 * 4096;
            int i = f >> 12;
            int rem = f & 4095;
            int e = rem >> 6, d = rem & 63;
            B2t[f] = (__bf16)m2b[i * 4096 + d * DD + e];
        }
    }
}

// ---------------------------------------------------------------------------
// Kernel 1: fused meta-projection, all matmuls on WMMA.
//   stage 1: h1 = relu(ste@W1+b1), bias = qkv@b2        (K=64 WMMA pairs)
//   stage 2: out[e,t] = sum_c h1[t,c] * (Bt_c @ qkv^T)  (K=4096, 128 WMMAs)
// grid = (621 token-tiles, 3 branches), block = 128 (4 waves)
// ---------------------------------------------------------------------------
__global__ __launch_bounds__(128)
void metaproj_kernel(const float* __restrict__ q_in,  const float* __restrict__ k_in,
                     const float* __restrict__ v_in,
                     const float* __restrict__ ste_q, const float* __restrict__ ste_k,
                     const float* __restrict__ ste_v,
                     const float* __restrict__ m1b,
                     const __bf16* __restrict__ Bt,  const __bf16* __restrict__ W1t,
                     const __bf16* __restrict__ B2t,
                     float* __restrict__ qkv_out)
{
    __shared__ float s_ste[16 * DD];
    __shared__ float s_qkv[16 * DD];
    __shared__ float s_h1[16 * DD];
    __shared__ float s_bias[16 * DD];

    const int i    = blockIdx.y;
    const int tok0 = blockIdx.x * 16;
    const int tid  = threadIdx.x;

    const float* ste = (i == 0) ? ste_q : (i == 1) ? ste_k : ste_v;
    const float* qkv = (i == 0) ? q_in  : (i == 1) ? k_in  : v_in;

#pragma unroll
    for (int u = 0; u < 8; ++u) {
        int idx = tid * 8 + u;
        s_ste[idx] = ste[(size_t)tok0 * DD + idx];
        s_qkv[idx] = qkv[(size_t)tok0 * DD + idx];
    }
    __syncthreads();

    const int lane = tid & 31;
    const int wave = tid >> 5;          // 0..3
    const int tl   = lane & 15;         // A-row / B-col low index
    const int hsel = lane >> 4;         // lane half
    const int khA  = hsel * 8;          // A K-offset: lanes>=16 hold K {8-15,24-31}
    const int khB  = hsel * 16;         // B K-offset: lanes>=16 hold K 16-31

    // ---- stage 1 (WMMA, K=64): h1 and bias tiles ----
    {
        v16bf ast[2], aq[2];
#pragma unroll
        for (int s = 0; s < 2; ++s)
#pragma unroll
            for (int j = 0; j < 16; ++j) {
                int kl = j + khB;
                ast[s][j] = (__bf16)s_ste[tl * DD + s * 32 + kl];
                aq[s][j]  = (__bf16)s_qkv[tl * DD + s * 32 + kl];
            }
        const int ccol = wave * 16 + tl;                 // output column (c or e)
        const __bf16* w1row = W1t + (size_t)(i * DD + ccol) * DD;
        const __bf16* b2row = B2t + (size_t)(i * DD + ccol) * DD;
        v16bf bw[2], bb[2];
#pragma unroll
        for (int s = 0; s < 2; ++s) {
            bw[s] = cat16(*(const v8bf*)(w1row + s * 32 + khB),
                          *(const v8bf*)(w1row + s * 32 + khB + 8));
            bb[s] = cat16(*(const v8bf*)(b2row + s * 32 + khB),
                          *(const v8bf*)(b2row + s * 32 + khB + 8));
        }
        v8f y = {};
        y = __builtin_amdgcn_wmma_f32_16x16x32_bf16(false, ast[0], false, bw[0], (short)0, y, false, false);
        y = __builtin_amdgcn_wmma_f32_16x16x32_bf16(false, ast[1], false, bw[1], (short)0, y, false, false);
        v8f yb = {};
        yb = __builtin_amdgcn_wmma_f32_16x16x32_bf16(false, aq[0], false, bb[0], (short)0, yb, false, false);
        yb = __builtin_amdgcn_wmma_f32_16x16x32_bf16(false, aq[1], false, bb[1], (short)0, yb, false, false);
        float b1v = m1b[i * DD + ccol];
#pragma unroll
        for (int r = 0; r < 8; ++r) {
            int trow = r + hsel * 8;                     // D rows: lanes>=16 hold M=r+8
            float h = y[r] + b1v;
            s_h1[trow * DD + ccol]   = h > 0.f ? h : 0.f;
            s_bias[trow * DD + ccol] = yb[r];
        }
    }
    __syncthreads();

    // ---- stage 2 (WMMA, K=4096): D[e,t] accumulated with per-c h1 scaling ----
    // B operand = qkv^T (d x tokens), loop-invariant
    v16bf bq[2];
#pragma unroll
    for (int s = 0; s < 2; ++s)
#pragma unroll
        for (int j = 0; j < 16; ++j)
            bq[s][j] = (__bf16)s_qkv[tl * DD + s * 32 + khA + ((j < 8) ? j : j + 8)];

    const int e0 = wave * 16;
    const __bf16* brow = Bt + (size_t)(i * DD + e0 + tl) * KDIM;  // A row = output col e

    auto loadA = [&](v8bf* dst, int c) {
        const __bf16* p = brow + c * DD + khA;
        dst[0] = *(const v8bf*)(p);
        dst[1] = *(const v8bf*)(p + 16);
        dst[2] = *(const v8bf*)(p + 32);
        dst[3] = *(const v8bf*)(p + 48);
    };

    v8f acc = {};
    v8bf buf[2][4];
    loadA(buf[0], 0);
#pragma unroll 2
    for (int c = 0; c < 64; ++c) {
        v8bf* cur = buf[c & 1];
        v8bf* nxt = buf[(c + 1) & 1];
        if (c < 63) loadA(nxt, c + 1);                   // double-buffered A stream
        __builtin_prefetch(brow + ((c + 6 < 64) ? (c + 6) : 63) * DD + khA, 0, 1);
        v16bf A0 = cat16(cur[0], cur[1]);
        v16bf A1 = cat16(cur[2], cur[3]);
        v8f y = {};
        y = __builtin_amdgcn_wmma_f32_16x16x32_bf16(false, A0, false, bq[0], (short)0, y, false, false);
        y = __builtin_amdgcn_wmma_f32_16x16x32_bf16(false, A1, false, bq[1], (short)0, y, false, false);
        float h1v = s_h1[tl * DD + c];                   // token t = tl (B-col lane index)
#pragma unroll
        for (int r = 0; r < 8; ++r) acc[r] = fmaf(h1v, y[r], acc[r]);
    }

    // ---- epilogue: rows are e = e0 + r + 8*hsel, column token t = tl ----
    {
        float4 o0, o1;
        const int ebase = e0 + hsel * 8;
        o0.x = acc[0] + s_bias[tl * DD + ebase + 0];
        o0.y = acc[1] + s_bias[tl * DD + ebase + 1];
        o0.z = acc[2] + s_bias[tl * DD + ebase + 2];
        o0.w = acc[3] + s_bias[tl * DD + ebase + 3];
        o1.x = acc[4] + s_bias[tl * DD + ebase + 4];
        o1.y = acc[5] + s_bias[tl * DD + ebase + 5];
        o1.z = acc[6] + s_bias[tl * DD + ebase + 6];
        o1.w = acc[7] + s_bias[tl * DD + ebase + 7];
        float* dst = qkv_out + (size_t)i * NTOK * DD + (size_t)(tok0 + tl) * DD + ebase;
        *(float4*)(dst)     = o0;
        *(float4*)(dst + 4) = o1;
    }
}

// ---------------------------------------------------------------------------
// Kernel 2: causal multi-head attention (T=12, H=8, DH=8) + final @wo.
// One 96-thread block per (b,n); thread = (head, t_q).
// ---------------------------------------------------------------------------
__global__ __launch_bounds__(96)
void attention_kernel(const float* __restrict__ qkvbuf, const float* __restrict__ wo,
                      float* __restrict__ out, float* __restrict__ att_out)
{
    __shared__ float s_k[TT * DD];
    __shared__ float s_v[TT * DD];
    __shared__ float s_pre[TT * DD];

    const int bn  = blockIdx.x;
    const int b   = bn / NN;
    const int n   = bn % NN;
    const int tid = threadIdx.x;

    const float* qb = qkvbuf;
    const float* kb = qkvbuf + (size_t)NTOK * DD;
    const float* vb = qkvbuf + (size_t)2 * NTOK * DD;

#pragma unroll
    for (int u = 0; u < 8; ++u) {
        int idx = tid * 8 + u;
        int t = idx >> 6, e = idx & 63;
        size_t g = ((size_t)(b * TT + t) * NN + n) * DD + e;
        s_k[idx] = kb[g];
        s_v[idx] = vb[g];
    }
    __syncthreads();

    const int h  = tid / TT;
    const int tq = tid % TT;
    float qv[DH];
    {
        size_t g = ((size_t)(b * TT + tq) * NN + n) * DD + h * DH;
#pragma unroll
        for (int j = 0; j < DH; ++j) qv[j] = qb[g + j];
    }

    const float scale = 0.35355339059327379f;   // 1/sqrt(8)
    float sc[TT];
    float mx = -1e30f;
    for (int tk = 0; tk < TT; ++tk) {
        if (tk <= tq) {
            float s = 0.f;
#pragma unroll
            for (int j = 0; j < DH; ++j) s += qv[j] * s_k[tk * DD + h * DH + j];
            s *= scale;
            sc[tk] = s;
            mx = s > mx ? s : mx;
        } else sc[tk] = 0.f;
    }
    float denom = 0.f;
    for (int tk = 0; tk <= tq; ++tk) { float e = __expf(sc[tk] - mx); sc[tk] = e; denom += e; }
    float inv = 1.f / denom;

    float o[DH];
#pragma unroll
    for (int j = 0; j < DH; ++j) o[j] = 0.f;
    for (int tk = 0; tk <= tq; ++tk) {
        float a = sc[tk] * inv;
        sc[tk] = a;
#pragma unroll
        for (int j = 0; j < DH; ++j) o[j] += a * s_v[tk * DD + h * DH + j];
    }
    for (int tk = tq + 1; tk < TT; ++tk) sc[tk] = 0.f;

    {   // att row: ((h*(B*N)+bn)*T + tq)*T + tk
        size_t abase = (((size_t)h * (BB * NN) + bn) * TT + tq) * TT;
        for (int tk = 0; tk < TT; ++tk) att_out[abase + tk] = sc[tk];
    }

#pragma unroll
    for (int j = 0; j < DH; ++j) s_pre[tq * DD + h * DH + j] = o[j];
    __syncthreads();

#pragma unroll
    for (int u = 0; u < 8; ++u) {
        int idx = tid * 8 + u;
        int t = idx >> 6, e = idx & 63;
        float s = 0.f;
        for (int f = 0; f < DD; ++f) s += s_pre[t * DD + f] * wo[f * DD + e];
        out[((size_t)(b * TT + t) * NN + n) * DD + e] = s;
    }
}

// ---------------------------------------------------------------------------
// Host launcher
// ---------------------------------------------------------------------------
extern "C" void kernel_launch(void* const* d_in, const int* in_sizes, int n_in,
                              void* d_out, int out_size, void* d_ws, size_t ws_size,
                              hipStream_t stream) {
    const float* queries = (const float*)d_in[0];
    const float* keys    = (const float*)d_in[1];
    const float* values  = (const float*)d_in[2];
    // d_in[3] = tms (unused)
    const float* ste_q   = (const float*)d_in[4];
    const float* ste_k   = (const float*)d_in[5];
    const float* ste_v   = (const float*)d_in[6];
    const float* meta1_w = (const float*)d_in[7];
    const float* meta1_b = (const float*)d_in[8];
    const float* meta2_w = (const float*)d_in[9];
    const float* meta2_b = (const float*)d_in[10];
    const float* wo      = (const float*)d_in[11];

    // workspace: [Bt 1.5MB][W1t 24KB][B2t 24KB][qkv f32 7.3MB]
    char*   ws     = (char*)d_ws;
    __bf16* Bt     = (__bf16*)ws;
    __bf16* W1t    = (__bf16*)(ws + (size_t)3 * M2ELEMS * sizeof(__bf16));
    __bf16* B2t    = (__bf16*)(ws + (size_t)3 * M2ELEMS * sizeof(__bf16) + 3 * 4096 * sizeof(__bf16));
    float*  qkvbuf = (float*)(ws + (size_t)3 * M2ELEMS * sizeof(__bf16) + 2 * 3 * 4096 * sizeof(__bf16));

    float* out     = (float*)d_out;
    float* att_out = out + (size_t)NTOK * DD;   // tuple order: out, att

    // 0) weights -> bf16 transposed ((786432+12288+12288)/2048 = 396 blocks)
    convert_kernel<<<dim3(396), dim3(256), 0, stream>>>(meta2_w, meta1_w, meta2_b, Bt, W1t, B2t);

    // 1) fused meta-projection (WMMA)
    metaproj_kernel<<<dim3(NTILE, 3), dim3(128), 0, stream>>>(
        queries, keys, values, ste_q, ste_k, ste_v,
        meta1_b, Bt, W1t, B2t, qkvbuf);

    // 2) attention + output projection
    attention_kernel<<<dim3(BB * NN), dim3(96), 0, stream>>>(qkvbuf, wo, out, att_out);
}
